// EgocentricHiveMind_41326175322215
// MI455X (gfx1250) — matrix operand order repaired and checked by reading, not compiled, not run on hardware
//
#include <hip/hip_runtime.h>
#include <hip/hip_bf16.h>
#include <stdint.h>

// ---------------------------------------------------------------------------
// EgocentricHiveMind forward for MI455X (gfx1250, wave32, WMMA).
// GEMMs: f16 WMMA (v_wmma_f32_16x16x32_f16) with f32 accumulate, 64x64 block
// tiles, async global->LDS staging (ASYNCcnt).  Entity MLPs algebraically
// split so the broadcast x_self part is computed once per type.
// ---------------------------------------------------------------------------

typedef __attribute__((ext_vector_type(16))) _Float16 v16h;
typedef __attribute__((ext_vector_type(8)))  float    v8f;

#define USE_ASYNC_LDS 1

#define T_DIM 64
#define B_DIM 256
#define TB    (T_DIM * B_DIM)          // 16384
#define N_ENT 16
#define M_ENT 33                        // 1 + 2*16
#define MROWS ((size_t)TB * M_ENT)      // 540672
#define XSELF 574
#define KSELF 576                       // padded to mult of 32
#define EF    64
#define KPOOL 640                       // 638 padded
#define LIN   128
#define LH    128

// ---- workspace layout (byte offsets, all naturally 256B aligned) ----------
constexpr size_t OFF_WSELF = 0;                                    // f16 576x64
constexpr size_t OFF_WSEB  = OFF_WSELF + (size_t)KSELF*EF*2;       // f16 576x64
constexpr size_t OFF_WSEA  = OFF_WSEB  + (size_t)KSELF*EF*2;       // f16 576x64
constexpr size_t OFF_WXEB  = OFF_WSEA  + (size_t)KSELF*EF*2;       // f16 8x64
constexpr size_t OFF_WXEA  = OFF_WXEB  + 8*EF*2;                   // f16 8x64
constexpr size_t OFF_WIN   = OFF_WXEA  + 8*EF*2;                   // f16 64x192
constexpr size_t OFF_WOUT  = OFF_WIN   + 64*192*2;                 // f16 64x64
constexpr size_t OFF_WSA   = OFF_WOUT  + 64*64*2;                  // f16 64x64
constexpr size_t OFF_WPOOL = OFF_WSA   + 64*64*2;                  // f16 640x128
constexpr size_t OFF_WIHT  = OFF_WPOOL + (size_t)KPOOL*LIN*2;      // f16 128x512
constexpr size_t OFF_WHHT  = OFF_WIHT  + 128*512*2;                // f16 128x512
constexpr size_t OFF_BCOMB = OFF_WHHT  + 128*512*2;                // f32 512
constexpr size_t OFF_XS    = OFF_BCOMB + 512*4;                    // f16 [TB,576]
constexpr size_t OFF_ZBB   = OFF_XS    + (size_t)TB*KSELF*2;       // f32 [TB,64]
constexpr size_t OFF_ZBA   = OFF_ZBB   + (size_t)TB*EF*4;          // f32 [TB,64]
constexpr size_t OFF_ZENT  = OFF_ZBA   + (size_t)TB*EF*4;          // f16 [TB,33,64]
constexpr size_t OFF_QKV   = OFF_ZENT  + MROWS*EF*2;               // f16 [MROWS,192]
constexpr size_t OFF_OTMP  = OFF_QKV;                              // alias (QKV dead)
constexpr size_t OFF_OATTN = OFF_QKV   + MROWS*192*2;              // f16 [MROWS,64]
constexpr size_t OFF_P     = OFF_OATTN + MROWS*EF*2;               // f16 [TB,640]
constexpr size_t OFF_ZLIN  = OFF_P     + (size_t)TB*KPOOL*2;       // f16 [TB,128]
constexpr size_t OFF_GPRE  = OFF_ZLIN  + (size_t)TB*LIN*2;         // f32 [TB,512]
constexpr size_t OFF_ZLSTM = OFF_GPRE  + (size_t)TB*512*4;         // f32 [TB,128]
constexpr size_t WS_NEED   = OFF_ZLSTM + (size_t)TB*LH*4;

// ---- GEMM epilogue flags --------------------------------------------------
#define GF_RELU   1
#define GF_RESID  2
#define GF_OUTF32 4

// ===========================================================================
// Generic WMMA GEMM:  C[M,N](ldc) = epi( A[M,K](lda,f16) @ W[K,N](f16) + bias )
// K % 32 == 0, N % 64 == 0, M % 64 == 0.  Block = 128 thr (4 waves),
// tile = 64 rows x 64 cols; wave w owns rows [16w,16w+16) x 64 cols
// (4 accumulators, A fragment reused 4x, 16 WMMA per block per K-step).
// ===========================================================================
__global__ __launch_bounds__(128)
void gemm_wmma_kernel(const _Float16* __restrict__ A,
                      const _Float16* __restrict__ W,
                      const float*    __restrict__ bias,
                      const _Float16* __restrict__ resid,
                      void* __restrict__ Cout,
                      int M, int N, int K, int lda, int ldc, int flags)
{
    __shared__ __align__(16) _Float16 lA[64 * 32];   // A tile, row-major [row][k]
    __shared__ __align__(16) _Float16 lBt[64 * 32];  // W tile, transposed [col][k]

    const int tid  = threadIdx.x;
    const int lane = tid & 31;
    const int wave = tid >> 5;
    const int m0   = blockIdx.x * 64;
    const int n0   = blockIdx.y * 64;

    v8f acc0 = {0.f,0.f,0.f,0.f,0.f,0.f,0.f,0.f};
    v8f acc1 = acc0, acc2 = acc0, acc3 = acc0;

    const int r_a   = wave * 16 + (lane & 15);  // A fragment row within tile
    const int kb4   = (lane >> 4) * 8;          // A frag u32 k-offset (koff/2)
    const int bcl   = lane & 15;                // B fragment column within group
    const int koffB = (lane >> 4) * 16;         // B frag k-offset (halves)

    // staging thread mappings (fixed per thread)
    const int a_i0 = tid * 16;                  // A: 16 contiguous halves
    const int a_r  = a_i0 >> 5, a_c = a_i0 & 31;
    const int b_j   = tid & 15;                 // B: k-pair index (k = 2j, 2j+1)
    const int b_cg  = tid >> 4;                 // B: column octet (col0 = 8*cg)

    for (int kb = 0; kb < K; kb += 32) {
        // ---- stage A tile 64x32 via async global->LDS (ASYNCcnt) ----------
        {
            const _Float16* gsrc = A + (size_t)(m0 + a_r) * lda + kb + a_c;
#if USE_ASYNC_LDS
            uint32_t ldst = (uint32_t)(uintptr_t)&lA[a_i0];
            uint64_t gptr = (uint64_t)(uintptr_t)gsrc;
            asm volatile("global_load_async_to_lds_b128 %0, %1, off"
                         :: "v"(ldst), "v"(gptr) : "memory");
            asm volatile("global_load_async_to_lds_b128 %0, %1, off offset:16"
                         :: "v"(ldst), "v"(gptr) : "memory");
#else
            *(uint4*)&lA[a_i0]     = *(const uint4*)gsrc;
            *(uint4*)&lA[a_i0 + 8] = *(const uint4*)(gsrc + 8);
#endif
        }
        // ---- stage W tile 32x64 -> transposed LDS [col][k], b32 packed ----
        {
            const _Float16* s0 = W + (size_t)(kb + 2 * b_j) * N + n0 + 8 * b_cg;
            const _Float16* s1 = s0 + N;
            union { uint4 q; _Float16 h[8]; } r0, r1;
            r0.q = *(const uint4*)s0;
            r1.q = *(const uint4*)s1;
            uint32_t* lB32 = (uint32_t*)lBt;
            union { _Float16 h[2]; uint32_t u; } p;
#pragma unroll
            for (int c = 0; c < 8; ++c) {
                p.h[0] = r0.h[c]; p.h[1] = r1.h[c];
                lB32[(8 * b_cg + c) * 16 + b_j] = p.u;
            }
        }
#if USE_ASYNC_LDS
        asm volatile("s_wait_asynccnt 0x0" ::: "memory");
#endif
        __syncthreads();

        // ---- fragments (k = 16*(lane/16) + s in both A and B) -------------------------------
        union { uint32_t u[8]; uint4 q[2]; v16h v; } af, bf0, bf1, bf2, bf3;
        const uint32_t* lAu = (const uint32_t*)lA;   // [64][16] k-pairs
        {
            const uint4* ap = (const uint4*)&lAu[r_a * 16 + kb4];
            af.q[0] = ap[0];                          // K = koff + 0..7
            af.q[1] = *(const uint4*)&lAu[r_a * 16 + 4 + kb4]; // K = koff+8..15
        }
        {
            const uint4* bp0 = (const uint4*)&lBt[(0 * 16 + bcl) * 32 + koffB];
            const uint4* bp1 = (const uint4*)&lBt[(1 * 16 + bcl) * 32 + koffB];
            const uint4* bp2 = (const uint4*)&lBt[(2 * 16 + bcl) * 32 + koffB];
            const uint4* bp3 = (const uint4*)&lBt[(3 * 16 + bcl) * 32 + koffB];
            bf0.q[0] = bp0[0]; bf0.q[1] = bp0[1];
            bf1.q[0] = bp1[0]; bf1.q[1] = bp1[1];
            bf2.q[0] = bp2[0]; bf2.q[1] = bp2[1];
            bf3.q[0] = bp3[0]; bf3.q[1] = bp3[1];
        }
        acc0 = __builtin_amdgcn_wmma_f32_16x16x32_f16(
                   false, af.v, false, bf0.v, (short)0, acc0, false, false);
        acc1 = __builtin_amdgcn_wmma_f32_16x16x32_f16(
                   false, af.v, false, bf1.v, (short)0, acc1, false, false);
        acc2 = __builtin_amdgcn_wmma_f32_16x16x32_f16(
                   false, af.v, false, bf2.v, (short)0, acc2, false, false);
        acc3 = __builtin_amdgcn_wmma_f32_16x16x32_f16(
                   false, af.v, false, bf3.v, (short)0, acc3, false, false);
        __syncthreads();
    }

    // ---- epilogue ----------------------------------------------------------
    const int rofs = (lane >> 4) << 3;
    const int rowb = m0 + wave * 16 + rofs;
#pragma unroll
    for (int g = 0; g < 4; ++g) {
        v8f a = (g == 0) ? acc0 : (g == 1) ? acc1 : (g == 2) ? acc2 : acc3;
        const int colg = n0 + g * 16 + (lane & 15);
        float bv = bias ? bias[colg] : 0.f;
#pragma unroll
        for (int i = 0; i < 8; ++i) {
            float v = a[i] + bv;
            if (flags & GF_RELU) v = fmaxf(v, 0.f);
            size_t idx = (size_t)(rowb + i) * ldc + colg;
            if (flags & GF_RESID) v += (float)resid[idx];
            if (flags & GF_OUTF32) ((float*)Cout)[idx] = v;
            else                   ((_Float16*)Cout)[idx] = (_Float16)v;
        }
    }
}

// ===========================================================================
// Weight prep: fold self_w halves, split entity weights, transpose LSTM
// weights, pad K dims to mult-of-32, convert to f16.
// ===========================================================================
constexpr int PN_WSELF = KSELF * EF;      // 36864
constexpr int PN_WSE   = KSELF * EF;
constexpr int PN_WXE   = 8 * EF;          // 512
constexpr int PN_WIN   = 64 * 192;
constexpr int PN_W64   = 64 * 64;
constexpr int PN_WPOOL = KPOOL * LIN;     // 81920
constexpr int PN_WLSTM = 128 * 512;
constexpr int PREP_TOTAL = PN_WSELF + 2*PN_WSE + 2*PN_WXE + PN_WIN + 2*PN_W64
                         + PN_WPOOL + 2*PN_WLSTM + 512;

__global__ void prep_weights_kernel(
    const float* __restrict__ self_w, const float* __restrict__ box_w,
    const float* __restrict__ agent_w, const float* __restrict__ in_w,
    const float* __restrict__ out_w, const float* __restrict__ sa_w,
    const float* __restrict__ pool_w, const float* __restrict__ w_ih,
    const float* __restrict__ w_hh, const float* __restrict__ b_ih,
    const float* __restrict__ b_hh, char* __restrict__ ws)
{
    _Float16* wself = (_Float16*)(ws + OFF_WSELF);
    _Float16* wseb  = (_Float16*)(ws + OFF_WSEB);
    _Float16* wsea  = (_Float16*)(ws + OFF_WSEA);
    _Float16* wxeb  = (_Float16*)(ws + OFF_WXEB);
    _Float16* wxea  = (_Float16*)(ws + OFF_WXEA);
    _Float16* win   = (_Float16*)(ws + OFF_WIN);
    _Float16* wout  = (_Float16*)(ws + OFF_WOUT);
    _Float16* wsa   = (_Float16*)(ws + OFF_WSA);
    _Float16* wpool = (_Float16*)(ws + OFF_WPOOL);
    _Float16* wiht  = (_Float16*)(ws + OFF_WIHT);
    _Float16* whht  = (_Float16*)(ws + OFF_WHHT);
    float*    bcomb = (float*)   (ws + OFF_BCOMB);

    for (int idx = blockIdx.x * blockDim.x + threadIdx.x; idx < PREP_TOTAL;
         idx += gridDim.x * blockDim.x) {
        int i = idx;
        if (i < PN_WSELF) { int k = i >> 6, n = i & 63;
            wself[i] = (_Float16)(k < XSELF ? self_w[k*EF+n] + self_w[(XSELF+k)*EF+n] : 0.f);
            continue; }
        i -= PN_WSELF;
        if (i < PN_WSE) { int k = i >> 6, n = i & 63;
            wseb[i] = (_Float16)(k < XSELF ? box_w[(8+k)*EF+n] : 0.f); continue; }
        i -= PN_WSE;
        if (i < PN_WSE) { int k = i >> 6, n = i & 63;
            wsea[i] = (_Float16)(k < XSELF ? agent_w[(8+k)*EF+n] : 0.f); continue; }
        i -= PN_WSE;
        if (i < PN_WXE) { wxeb[i] = (_Float16)box_w[i];   continue; }
        i -= PN_WXE;
        if (i < PN_WXE) { wxea[i] = (_Float16)agent_w[i]; continue; }
        i -= PN_WXE;
        if (i < PN_WIN) { win[i] = (_Float16)in_w[i];  continue; }
        i -= PN_WIN;
        if (i < PN_W64) { wout[i] = (_Float16)out_w[i]; continue; }
        i -= PN_W64;
        if (i < PN_W64) { wsa[i] = (_Float16)sa_w[i];  continue; }
        i -= PN_W64;
        if (i < PN_WPOOL) { int k = i >> 7, n = i & 127;
            wpool[i] = (_Float16)(k < 638 ? pool_w[k*LIN+n] : 0.f); continue; }
        i -= PN_WPOOL;
        if (i < PN_WLSTM) { int k = i >> 9, n = i & 511;
            wiht[i] = (_Float16)w_ih[n*128 + k]; continue; }
        i -= PN_WLSTM;
        if (i < PN_WLSTM) { int k = i >> 9, n = i & 511;
            whht[i] = (_Float16)w_hh[n*128 + k]; continue; }
        i -= PN_WLSTM;
        if (i < 512) { bcomb[i] = b_ih[i] + b_hh[i]; continue; }
    }
}

// ===========================================================================
// Lidar conv (VALID, 9 filters, k=3) + concat -> x_self (f16, padded to 576)
// ===========================================================================
__global__ void xself_kernel(const float* __restrict__ x_agent,
                             const float* __restrict__ x_lidar,
                             const float* __restrict__ conv_w,
                             const float* __restrict__ conv_b,
                             _Float16* __restrict__ Xs)
{
    int idx = blockIdx.x * blockDim.x + threadIdx.x;
    if (idx >= TB * KSELF) return;
    int c = idx % KSELF, row = idx / KSELF;
    float v;
    if (c < 16) v = x_agent[row * 16 + c];
    else if (c < XSELF) {
        int cc = c - 16, f = cc / 62, p = cc % 62;
        float a = conv_b[f];
#pragma unroll
        for (int k = 0; k < 3; ++k) a += x_lidar[row * 64 + p + k] * conv_w[f * 3 + k];
        v = fmaxf(a, 0.f);
    } else v = 0.f;
    Xs[(size_t)row * KSELF + c] = (_Float16)v;
}

// ===========================================================================
// Entity fixup: z_ent[row, 1+e2, n] = relu(zbase + x_e[8] . wxe[:,n])
// ===========================================================================
__global__ void entity_kernel(const float* __restrict__ xe_box,
                              const float* __restrict__ xe_agent,
                              const float* __restrict__ zb_box,
                              const float* __restrict__ zb_agent,
                              const _Float16* __restrict__ wxe_box,
                              const _Float16* __restrict__ wxe_agent,
                              _Float16* __restrict__ zent)
{
    int idx = blockIdx.x * blockDim.x + threadIdx.x;
    if (idx >= TB * 32 * EF) return;
    int n   = idx & 63;
    int e2  = (idx >> 6) & 31;
    int row = idx >> 11;
    bool ag = e2 >= 16;
    int  e  = e2 & 15;
    const float*    xe  = (ag ? xe_agent : xe_box) + ((size_t)row * 16 + e) * 8;
    const _Float16* wxe = ag ? wxe_agent : wxe_box;
    float acc = (ag ? zb_agent : zb_box)[(size_t)row * EF + n];
#pragma unroll
    for (int j = 0; j < 8; ++j) acc += xe[j] * (float)wxe[j * EF + n];
    zent[((size_t)row * M_ENT + 1 + e2) * EF + n] = (_Float16)fmaxf(acc, 0.f);
}

// ===========================================================================
// Attention: per (t,b) block; QKV -> LDS; thread = (head, query-row)
// ===========================================================================
__global__ __launch_bounds__(160)
void attn_kernel(const _Float16* __restrict__ qkv, _Float16* __restrict__ oattn)
{
    __shared__ float sQ[M_ENT * EF], sK[M_ENT * EF], sV[M_ENT * EF];
    int blk = blockIdx.x;
    const _Float16* src = qkv + (size_t)blk * M_ENT * 192;
    for (int i = threadIdx.x; i < M_ENT * EF; i += blockDim.x) {
        int m = i >> 6, c = i & 63;
        sQ[i] = (float)src[m * 192 + c];
        sK[i] = (float)src[m * 192 + 64 + c];
        sV[i] = (float)src[m * 192 + 128 + c];
    }
    __syncthreads();
    int w = threadIdx.x;
    if (w < 4 * M_ENT) {
        int h = w / M_ENT, m = w % M_ENT;
        float s[M_ENT];
        float mx = -1e30f;
#pragma unroll 1
        for (int n = 0; n < M_ENT; ++n) {
            float d = 0.f;
#pragma unroll
            for (int dd = 0; dd < 16; ++dd)
                d += sQ[m * EF + h * 16 + dd] * sK[n * EF + h * 16 + dd];
            s[n] = d * 0.25f;                    // 1/sqrt(16)
            mx = fmaxf(mx, s[n]);
        }
        float sum = 0.f;
        for (int n = 0; n < M_ENT; ++n) { s[n] = __expf(s[n] - mx); sum += s[n]; }
        float inv = 1.f / sum;
#pragma unroll 1
        for (int dd = 0; dd < 16; ++dd) {
            float o = 0.f;
            for (int n = 0; n < M_ENT; ++n) o += s[n] * sV[n * EF + h * 16 + dd];
            oattn[((size_t)blk * M_ENT + m) * EF + h * 16 + dd] = (_Float16)(o * inv);
        }
    }
}

// ===========================================================================
// Mean-pool over 33 entities + concat(x_self, z_pool) (f16, padded to 640)
// ===========================================================================
__global__ void pool_kernel(const _Float16* __restrict__ Xs,
                            const _Float16* __restrict__ zent,
                            _Float16* __restrict__ P)
{
    int idx = blockIdx.x * blockDim.x + threadIdx.x;
    if (idx >= TB * KPOOL) return;
    int c = idx % KPOOL, row = idx / KPOOL;
    _Float16 v;
    if (c < XSELF) v = Xs[(size_t)row * KSELF + c];
    else if (c < 638) {
        int n = c - XSELF;
        float s = 0.f;
        for (int m = 0; m < M_ENT; ++m) s += (float)zent[((size_t)row * M_ENT + m) * EF + n];
        v = (_Float16)(s * (1.f / 33.f));
    } else v = (_Float16)0.f;
    P[(size_t)row * KPOOL + c] = v;
}

// ===========================================================================
// Masked LSTM: one block per batch element, h/c/gates in LDS.
// gpre already holds z@w_ih^T + (b_ih+b_hh).
// ===========================================================================
__global__ __launch_bounds__(512)
void lstm_kernel(const float* __restrict__ gpre, const float* __restrict__ done,
                 const _Float16* __restrict__ whhT,
                 const float* __restrict__ h0, const float* __restrict__ c0,
                 float* __restrict__ zlstm)
{
    __shared__ float sh[LH], sc[LH], sg[512];
    int b = blockIdx.x, tid = threadIdx.x;
    if (tid < LH) { sh[tid] = h0[b * LH + tid]; sc[tid] = c0[b * LH + tid]; }
    __syncthreads();
    for (int t = 0; t < T_DIM; ++t) {
        float m = 1.f - done[t * B_DIM + b];
        if (tid < LH) { sh[tid] *= m; sc[tid] *= m; }
        __syncthreads();
        float acc = gpre[((size_t)t * B_DIM + b) * 512 + tid];
#pragma unroll 4
        for (int k = 0; k < LH; ++k) acc += sh[k] * (float)whhT[k * 512 + tid];
        sg[tid] = acc;
        __syncthreads();
        if (tid < LH) {
            float gi = sg[tid], gf = sg[128 + tid], gg = sg[256 + tid], go = sg[384 + tid];
            float i_ = 1.f / (1.f + __expf(-gi));
            float f_ = 1.f / (1.f + __expf(-gf));
            float g_ = tanhf(gg);
            float o_ = 1.f / (1.f + __expf(-go));
            float c  = f_ * sc[tid] + i_ * g_;
            float h  = o_ * tanhf(c);
            sc[tid] = c; sh[tid] = h;
            zlstm[((size_t)t * B_DIM + b) * LH + tid] = h;
        }
        __syncthreads();
    }
}

// ===========================================================================
// Heads: log_softmax + entropy + action gather. d_out = [actions|logp|entr].
// ===========================================================================
__device__ __forceinline__
void head_eval(const float* z, const float* w, const float* b, int C,
               int act, float* lp, float* ent)
{
    float l[3];
    for (int c = 0; c < C; ++c) {
        float a = b[c];
        for (int k = 0; k < LH; ++k) a += z[k] * w[k * C + c];
        l[c] = a;
    }
    float mx = l[0];
    for (int c = 1; c < C; ++c) mx = fmaxf(mx, l[c]);
    float s = 0.f;
    for (int c = 0; c < C; ++c) s += __expf(l[c] - mx);
    float lse = mx + __logf(s);
    float e = 0.f;
    for (int c = 0; c < C; ++c) e += __expf(l[c] - lse) * l[c];
    *ent = lse - e;
    *lp  = l[act] - lse;
}

__global__ void heads_kernel(const float* __restrict__ zlstm,
                             const float* h0w, const float* h0b,
                             const float* h1w, const float* h1b,
                             const float* h2w, const float* h2b,
                             const int* __restrict__ actions,
                             float* __restrict__ out)
{
    int r = blockIdx.x * blockDim.x + threadIdx.x;
    if (r >= TB) return;
    const float* z = zlstm + (size_t)r * LH;
    const float* ws3[3] = {h0w, h1w, h2w};
    const float* bs3[3] = {h0b, h1b, h2b};
    const int    nc[3]  = {3, 3, 2};
    for (int hd = 0; hd < 3; ++hd) {
        int a = actions[r * 3 + hd];
        float lp, en;
        head_eval(z, ws3[hd], bs3[hd], nc[hd], a, &lp, &en);
        out[r * 3 + hd]              = (float)a;   // echoed actions
        out[TB * 3 + r * 3 + hd]     = lp;         // logprobs
        out[TB * 6 + r * 3 + hd]     = en;         // entropies
    }
}

// ===========================================================================
// Host launch
// ===========================================================================
#define WSP(T, off) ((T*)((char*)d_ws + (off)))

extern "C" void kernel_launch(void* const* d_in, const int* in_sizes, int n_in,
                              void* d_out, int out_size, void* d_ws, size_t ws_size,
                              hipStream_t stream) {
    if (ws_size < WS_NEED) return;

    const float* x_agent  = (const float*)d_in[0];
    const float* x_lidar  = (const float*)d_in[1];
    const float* xe_box   = (const float*)d_in[2];
    const float* xe_agent = (const float*)d_in[3];
    const float* done     = (const float*)d_in[4];
    const int*   actions  = (const int*)  d_in[5];
    const float* h0       = (const float*)d_in[6];
    const float* c0       = (const float*)d_in[7];
    const float* conv_w   = (const float*)d_in[8];
    const float* conv_b   = (const float*)d_in[9];
    const float* self_w   = (const float*)d_in[10];
    const float* self_b   = (const float*)d_in[11];
    const float* box_w    = (const float*)d_in[12];
    const float* box_b    = (const float*)d_in[13];
    const float* agent_w  = (const float*)d_in[14];
    const float* agent_b  = (const float*)d_in[15];
    const float* in_w     = (const float*)d_in[16];
    const float* in_b     = (const float*)d_in[17];
    const float* out_w    = (const float*)d_in[18];
    const float* out_b    = (const float*)d_in[19];
    const float* sa_w     = (const float*)d_in[20];
    const float* sa_b     = (const float*)d_in[21];
    const float* pool_w   = (const float*)d_in[22];
    const float* pool_b   = (const float*)d_in[23];
    const float* w_ih     = (const float*)d_in[24];
    const float* w_hh     = (const float*)d_in[25];
    const float* b_ih     = (const float*)d_in[26];
    const float* b_hh     = (const float*)d_in[27];
    const float* head0_w  = (const float*)d_in[28];
    const float* head0_b  = (const float*)d_in[29];
    const float* head1_w  = (const float*)d_in[30];
    const float* head1_b  = (const float*)d_in[31];
    const float* head2_w  = (const float*)d_in[32];
    const float* head2_b  = (const float*)d_in[33];

    // 1) weight prep + x_self build
    prep_weights_kernel<<<(PREP_TOTAL + 255) / 256, 256, 0, stream>>>(
        self_w, box_w, agent_w, in_w, out_w, sa_w, pool_w, w_ih, w_hh,
        b_ih, b_hh, (char*)d_ws);
    xself_kernel<<<(TB * KSELF + 255) / 256, 256, 0, stream>>>(
        x_agent, x_lidar, conv_w, conv_b, WSP(_Float16, OFF_XS));

    // 2) z_self = relu(Xs @ wself + b) -> entity slot 0 of ZENT
    gemm_wmma_kernel<<<dim3(TB / 64, 1), 128, 0, stream>>>(
        WSP(_Float16, OFF_XS), WSP(_Float16, OFF_WSELF), self_b, nullptr,
        WSP(_Float16, OFF_ZENT), TB, EF, KSELF, KSELF, M_ENT * EF, GF_RELU);

    // 3) shared entity bases: zb = Xs @ wse + bias (f32, no relu)
    gemm_wmma_kernel<<<dim3(TB / 64, 1), 128, 0, stream>>>(
        WSP(_Float16, OFF_XS), WSP(_Float16, OFF_WSEB), box_b, nullptr,
        WSP(float, OFF_ZBB), TB, EF, KSELF, KSELF, EF, GF_OUTF32);
    gemm_wmma_kernel<<<dim3(TB / 64, 1), 128, 0, stream>>>(
        WSP(_Float16, OFF_XS), WSP(_Float16, OFF_WSEA), agent_b, nullptr,
        WSP(float, OFF_ZBA), TB, EF, KSELF, KSELF, EF, GF_OUTF32);

    // 4) per-entity rank-8 fixup + relu -> ZENT slots 1..32
    entity_kernel<<<(TB * 32 * EF + 255) / 256, 256, 0, stream>>>(
        xe_box, xe_agent, WSP(float, OFF_ZBB), WSP(float, OFF_ZBA),
        WSP(_Float16, OFF_WXEB), WSP(_Float16, OFF_WXEA),
        WSP(_Float16, OFF_ZENT));

    // 5) qkv = ZENT @ in_proj + b
    gemm_wmma_kernel<<<dim3(MROWS / 64, 3), 128, 0, stream>>>(
        WSP(_Float16, OFF_ZENT), WSP(_Float16, OFF_WIN), in_b, nullptr,
        WSP(_Float16, OFF_QKV), (int)MROWS, 192, EF, EF, 192, 0);

    // 6) attention
    attn_kernel<<<TB, 160, 0, stream>>>(
        WSP(_Float16, OFF_QKV), WSP(_Float16, OFF_OATTN));

    // 7) out_proj (OTMP aliases dead QKV region)
    gemm_wmma_kernel<<<dim3(MROWS / 64, 1), 128, 0, stream>>>(
        WSP(_Float16, OFF_OATTN), WSP(_Float16, OFF_WOUT), out_b, nullptr,
        WSP(_Float16, OFF_OTMP), (int)MROWS, EF, EF, EF, EF, 0);

    // 8) resid = ZENT + relu(OTMP @ sa_w + b)   (in place on ZENT)
    gemm_wmma_kernel<<<dim3(MROWS / 64, 1), 128, 0, stream>>>(
        WSP(_Float16, OFF_OTMP), WSP(_Float16, OFF_WSA), sa_b,
        WSP(_Float16, OFF_ZENT), WSP(_Float16, OFF_ZENT),
        (int)MROWS, EF, EF, EF, EF, GF_RELU | GF_RESID);

    // 9) mean pool + concat
    pool_kernel<<<(TB * KPOOL + 255) / 256, 256, 0, stream>>>(
        WSP(_Float16, OFF_XS), WSP(_Float16, OFF_ZENT), WSP(_Float16, OFF_P));

    // 10) z = relu(P @ pool_w + b)
    gemm_wmma_kernel<<<dim3(TB / 64, 2), 128, 0, stream>>>(
        WSP(_Float16, OFF_P), WSP(_Float16, OFF_WPOOL), pool_b, nullptr,
        WSP(_Float16, OFF_ZLIN), TB, LIN, KPOOL, KPOOL, LIN, GF_RELU);

    // 11) gpre = z @ w_ih^T + (b_ih + b_hh)
    gemm_wmma_kernel<<<dim3(TB / 64, 8), 128, 0, stream>>>(
        WSP(_Float16, OFF_ZLIN), WSP(_Float16, OFF_WIHT),
        WSP(float, OFF_BCOMB), nullptr,
        WSP(float, OFF_GPRE), TB, 512, LIN, LIN, 512, GF_OUTF32);

    // 12) masked LSTM scan
    lstm_kernel<<<B_DIM, 512, 0, stream>>>(
        WSP(float, OFF_GPRE), done, WSP(_Float16, OFF_WHHT), h0, c0,
        WSP(float, OFF_ZLSTM));

    // 13) heads
    heads_kernel<<<(TB + 255) / 256, 256, 0, stream>>>(
        WSP(float, OFF_ZLSTM), head0_w, head0_b, head1_w, head1_b,
        head2_w, head2_b, actions, (float*)d_out);
}
